// BitLinear_16716012716242
// MI455X (gfx1250) — matrix operand order repaired and checked by reading, not compiled, not run on hardware
//
#include <hip/hip_runtime.h>

typedef __attribute__((ext_vector_type(16))) __bf16 v16bf;
typedef __attribute__((ext_vector_type(8)))  float  v8f;

#define MDIM 16384
#define NDIM 4096
#define KDIM 4096

#define BM 128
#define BN 128
#define BK 32
#define PAD 8
#define LDT (BK + PAD)          // 40 halves per LDS row (80 B): b128 frag loads tile all 64 banks
#define NKT (KDIM / BK)         // 128 K-steps
#define RB  2048                // reduction blocks

// ---- CDNA5 async-to-LDS (gfx1250) -------------------------------------------
#if __has_builtin(__builtin_amdgcn_global_load_async_to_lds_b128) && \
    __has_builtin(__builtin_amdgcn_s_wait_asynccnt)
#define HAVE_ASYNC 1
#else
#define HAVE_ASYNC 0
#endif

#if HAVE_ASYNC
// builtin prototype: (v4i32 as(1)*, v4i32 as(3)*, imm offset, imm cpol)
typedef int v4i __attribute__((vector_size(4 * sizeof(int))));
typedef __attribute__((address_space(1))) v4i g_v4i;
typedef __attribute__((address_space(3))) v4i l_v4i;
__device__ __forceinline__ void async_copy_b128(const void* g, void* l) {
  // flat->as3 cast truncates to the 32-bit LDS offset (ISA 10.2 aperture rules)
  __builtin_amdgcn_global_load_async_to_lds_b128((g_v4i*)g, (l_v4i*)l, 0, 0);
}
#define NBUF 3                  // DMA runs two tiles ahead (async completes in order)
#else
#define NBUF 2
#endif

// float -> bf16, round-to-nearest-even (bit trick)
__device__ __forceinline__ unsigned short f2bf_rne(float f) {
  unsigned u = __float_as_uint(f);
  return (unsigned short)((u + 0x7FFFu + ((u >> 16) & 1u)) >> 16);
}

// ---------------- Pass 1: per-block |W| partial sums (deterministic) ----------------
__global__ __launch_bounds__(256) void absum_partial(const float* __restrict__ W,
                                                     float* __restrict__ part) {
  __shared__ float sm[256];
  const int tid = threadIdx.x;
  const int bid = blockIdx.x;
  float s = 0.0f;
  #pragma unroll 4
  for (int i = 0; i < 32; ++i) {
    size_t idx = (size_t)i * (RB * 256) + (size_t)bid * 256 + tid;
    s += fabsf(W[idx]);
  }
  sm[tid] = s;
  __syncthreads();
  for (int off = 128; off > 0; off >>= 1) {
    if (tid < off) sm[tid] += sm[tid + off];
    __syncthreads();
  }
  if (tid == 0) part[bid] = sm[0];
}

// ---------------- Pass 2: fold partials -> scale (deterministic) ----------------
__global__ __launch_bounds__(256) void absum_final(const float* __restrict__ part,
                                                   float* __restrict__ scaleOut) {
  __shared__ float sm[256];
  const int tid = threadIdx.x;
  float s = 0.0f;
  #pragma unroll
  for (int j = 0; j < RB / 256; ++j) s += part[j * 256 + tid];
  sm[tid] = s;
  __syncthreads();
  for (int off = 128; off > 0; off >>= 1) {
    if (tid < off) sm[tid] += sm[tid + off];
    __syncthreads();
  }
  if (tid == 0) scaleOut[0] = fmaxf(sm[0] * (1.0f / 16777216.0f), 1e-8f);
}

// ---------------- Pass 3a: X fp32 -> bf16 (RNE), 8 elems/thread ----------------
__global__ __launch_bounds__(256) void convert_x(const float* __restrict__ X,
                                                 unsigned short* __restrict__ Xb) {
  const size_t base = ((size_t)blockIdx.x * 256 + threadIdx.x) * 8;
  float4 a = *(const float4*)(X + base);
  float4 b = *(const float4*)(X + base + 4);
  uint4 o;
  o.x = (unsigned)f2bf_rne(a.x) | ((unsigned)f2bf_rne(a.y) << 16);
  o.y = (unsigned)f2bf_rne(a.z) | ((unsigned)f2bf_rne(a.w) << 16);
  o.z = (unsigned)f2bf_rne(b.x) | ((unsigned)f2bf_rne(b.y) << 16);
  o.w = (unsigned)f2bf_rne(b.z) | ((unsigned)f2bf_rne(b.w) << 16);
  *(uint4*)(Xb + base) = o;
}

// ---------------- Pass 3b: W -> ternary bf16 (exact), 8 elems/thread ----------------
__global__ __launch_bounds__(256) void quant_w(const float* __restrict__ W,
                                               const float* __restrict__ wsScale,
                                               unsigned short* __restrict__ Wq) {
  const float inv = 1.0f / wsScale[0];
  const size_t base = ((size_t)blockIdx.x * 256 + threadIdx.x) * 8;
  float4 a = *(const float4*)(W + base);
  float4 b = *(const float4*)(W + base + 4);
  float q[8] = {a.x, a.y, a.z, a.w, b.x, b.y, b.z, b.w};
  unsigned short h[8];
  #pragma unroll
  for (int i = 0; i < 8; ++i)
    h[i] = f2bf_rne(fminf(fmaxf(rintf(q[i] * inv), -1.0f), 1.0f));
  uint4 o;
  o.x = (unsigned)h[0] | ((unsigned)h[1] << 16);
  o.y = (unsigned)h[2] | ((unsigned)h[3] << 16);
  o.z = (unsigned)h[4] | ((unsigned)h[5] << 16);
  o.w = (unsigned)h[6] | ((unsigned)h[7] << 16);
  *(uint4*)(Wq + base) = o;
}

// ---------------- shared WMMA tile step & epilogue ----------------
__device__ __forceinline__ void wmma_step(const unsigned short (*__restrict__ Asb)[LDT],
                                          const unsigned short (*__restrict__ Bsb)[LDT],
                                          int am, int bn, int l16, int ak0, int bk0,
                                          v8f acc[4][2]) {
  union Frag { uint4 q[2]; v16bf v; };
  Frag fa[4], fb[2];
  #pragma unroll
  for (int mt = 0; mt < 4; ++mt) {
    const unsigned short* ap = &Asb[am + mt * 16 + l16][ak0];
    fa[mt].q[0] = *(const uint4*)(ap);
    fa[mt].q[1] = *(const uint4*)(ap + 8);
  }
  #pragma unroll
  for (int nt = 0; nt < 2; ++nt) {
    const unsigned short* bp = &Bsb[bn + nt * 16 + l16][bk0];
    fb[nt].q[0] = *(const uint4*)(bp);
    fb[nt].q[1] = *(const uint4*)(bp + 8);
  }
  #pragma unroll
  for (int mt = 0; mt < 4; ++mt)
    #pragma unroll
    for (int nt = 0; nt < 2; ++nt)
      acc[mt][nt] = __builtin_amdgcn_wmma_f32_16x16x32_bf16(
          false, fa[mt].v, false, fb[nt].v, (short)0, acc[mt][nt], false, false);
}

__device__ __forceinline__ void epilogue(float* __restrict__ Y,
                                         const float* __restrict__ bias, float scale,
                                         int m0, int n0, int am, int bn, int lhalf,
                                         int l16, v8f acc[4][2]) {
  #pragma unroll
  for (int nt = 0; nt < 2; ++nt) {
    const int gn = n0 + bn + nt * 16 + l16;
    const float bv = bias[gn];
    #pragma unroll
    for (int mt = 0; mt < 4; ++mt) {
      #pragma unroll
      for (int v = 0; v < 8; ++v) {
        const int gm = m0 + am + mt * 16 + lhalf * 8 + v;
        Y[(size_t)gm * NDIM + gn] = acc[mt][nt][v] * scale + bv;
      }
    }
  }
}

// ---------------- Pass 4 (fast path): pure-bf16 WMMA GEMM, async-to-LDS staging ----------------
__global__ __launch_bounds__(256) void bitlinear_gemm_pre(
    const unsigned short* __restrict__ Xb, const unsigned short* __restrict__ Wq,
    const float* __restrict__ bias, const float* __restrict__ wsScale,
    float* __restrict__ Y) {
  __shared__ __align__(16) unsigned short As[NBUF][BM][LDT];
  __shared__ __align__(16) unsigned short Bs[NBUF][BN][LDT];

  const int tid   = threadIdx.x;
  const int wid   = tid >> 5;
  const int lane  = tid & 31;
  const int lhalf = lane >> 4;
  const int l16   = lane & 15;

  const int m0 = blockIdx.y * BM;
  const int n0 = blockIdx.x * BN;
  const float scale = wsScale[0];

  // staging map: tile = 128 rows x 64 B = 512 b128 chunks; 2 chunks/thread/matrix
  const int r0 = tid >> 2;           // rows 0..63 (pass p adds 64)
  const int q0 = (tid & 3) * 8;      // bf16 col offset 0,8,16,24

  const unsigned short* xsrc = Xb + (size_t)(m0 + r0) * KDIM + q0;
  const unsigned short* wsrc = Wq + (size_t)(n0 + r0) * KDIM + q0;

  const int am  = (wid & 1) * 64;
  const int bn  = (wid >> 1) * 32;
  const int ak0 = lhalf * 16;
  const int bk0 = lhalf * 16;

  v8f zero = {};
  v8f acc[4][2];
  #pragma unroll
  for (int mt = 0; mt < 4; ++mt)
    #pragma unroll
    for (int nt = 0; nt < 2; ++nt) acc[mt][nt] = zero;

#if HAVE_ASYNC
  auto issue = [&](int kt, int buf) {
    const size_t ko = (size_t)kt * BK;
    #pragma unroll
    for (int p = 0; p < 2; ++p) {
      const int r = p * 64 + r0;
      async_copy_b128(xsrc + (size_t)(p * 64) * KDIM + ko, &As[buf][r][q0]);
      async_copy_b128(wsrc + (size_t)(p * 64) * KDIM + ko, &Bs[buf][r][q0]);
    }
  };

  // DMA depth 2: issue tiles 0 and 1, retire tile 0 only (async completes in order)
  issue(0, 0);
  issue(1, 1);
  __builtin_amdgcn_s_wait_asynccnt(4);
  __syncthreads();

  int cur = 0;
  #pragma unroll 1
  for (int kt = 0; kt < NKT; ++kt) {
    const int nb = (cur == NBUF - 1) ? 0 : cur + 1;   // buffer of tile kt+1
    const int ib = (nb == NBUF - 1) ? 0 : nb + 1;     // buffer of tile kt+2
    if (kt + 2 < NKT) issue(kt + 2, ib);              // DMA two tiles ahead
    wmma_step(As[cur], Bs[cur], am, bn, l16, ak0, bk0, acc);
    if (kt + 2 < NKT) {
      __builtin_amdgcn_s_wait_asynccnt(4);            // tile kt+1 landed; kt+2 in flight
    } else if (kt + 1 < NKT) {
      __builtin_amdgcn_s_wait_asynccnt(0);            // drain tail
    }
    __syncthreads();
    cur = nb;
  }
#else
  uint4 ra[2], rb[2];
  auto fetch = [&](int kt) {
    const size_t ko = (size_t)kt * BK;
    #pragma unroll
    for (int p = 0; p < 2; ++p) {
      ra[p] = *(const uint4*)(xsrc + (size_t)(p * 64) * KDIM + ko);
      rb[p] = *(const uint4*)(wsrc + (size_t)(p * 64) * KDIM + ko);
    }
  };
  auto stage = [&](int buf) {
    #pragma unroll
    for (int p = 0; p < 2; ++p) {
      const int r = p * 64 + r0;
      *(uint4*)(&As[buf][r][q0]) = ra[p];
      *(uint4*)(&Bs[buf][r][q0]) = rb[p];
    }
  };

  fetch(0);
  stage(0);
  __syncthreads();
  #pragma unroll 1
  for (int kt = 0; kt < NKT; ++kt) {
    const int cur = kt & 1;
    if (kt + 1 < NKT) fetch(kt + 1);
    wmma_step(As[cur], Bs[cur], am, bn, l16, ak0, bk0, acc);
    if (kt + 1 < NKT) stage(cur ^ 1);
    __syncthreads();
  }
#endif

  epilogue(Y, bias, scale, m0, n0, am, bn, lhalf, l16, acc);
}

// ---------------- Fallback: on-the-fly conversion GEMM (small workspace) ----------------
__global__ __launch_bounds__(256) void bitlinear_gemm_otf(
    const float* __restrict__ X, const float* __restrict__ W,
    const float* __restrict__ bias, const float* __restrict__ wsScale,
    float* __restrict__ Y) {
  __shared__ __align__(16) unsigned short As[2][BM][LDT];
  __shared__ __align__(16) unsigned short Bs[2][BN][LDT];

  const int tid   = threadIdx.x;
  const int wid   = tid >> 5;
  const int lane  = tid & 31;
  const int lhalf = lane >> 4;
  const int l16   = lane & 15;

  const int m0 = blockIdx.y * BM;
  const int n0 = blockIdx.x * BN;

  const float scale     = wsScale[0];
  const float inv_scale = 1.0f / scale;

  const int lr = tid >> 3;
  const int lc = (tid & 7) * 4;

  const float* xbase = X + (size_t)(m0 + lr) * KDIM + lc;
  const float* wbase = W + (size_t)(n0 + lr) * KDIM + lc;

  float4 ra[4], rb[4];
  auto fetch = [&](int kt) {
    const size_t ko = (size_t)kt * BK;
    #pragma unroll
    for (int p = 0; p < 4; ++p) {
      ra[p] = *(const float4*)(xbase + (size_t)(p * 32) * KDIM + ko);
      rb[p] = *(const float4*)(wbase + (size_t)(p * 32) * KDIM + ko);
    }
  };
  auto stage = [&](int buf) {
    #pragma unroll
    for (int p = 0; p < 4; ++p) {
      const int r = p * 32 + lr;
      uint2 pa;
      pa.x = (unsigned)f2bf_rne(ra[p].x) | ((unsigned)f2bf_rne(ra[p].y) << 16);
      pa.y = (unsigned)f2bf_rne(ra[p].z) | ((unsigned)f2bf_rne(ra[p].w) << 16);
      *(uint2*)(&As[buf][r][lc]) = pa;
      float q0 = fminf(fmaxf(rintf(rb[p].x * inv_scale), -1.0f), 1.0f);
      float q1 = fminf(fmaxf(rintf(rb[p].y * inv_scale), -1.0f), 1.0f);
      float q2 = fminf(fmaxf(rintf(rb[p].z * inv_scale), -1.0f), 1.0f);
      float q3 = fminf(fmaxf(rintf(rb[p].w * inv_scale), -1.0f), 1.0f);
      uint2 pb;
      pb.x = (unsigned)f2bf_rne(q0) | ((unsigned)f2bf_rne(q1) << 16);
      pb.y = (unsigned)f2bf_rne(q2) | ((unsigned)f2bf_rne(q3) << 16);
      *(uint2*)(&Bs[buf][r][lc]) = pb;
    }
  };

  const int am  = (wid & 1) * 64;
  const int bn  = (wid >> 1) * 32;
  const int ak0 = lhalf * 16;
  const int bk0 = lhalf * 16;

  v8f zero = {};
  v8f acc[4][2];
  #pragma unroll
  for (int mt = 0; mt < 4; ++mt)
    #pragma unroll
    for (int nt = 0; nt < 2; ++nt) acc[mt][nt] = zero;

  fetch(0);
  stage(0);
  __syncthreads();
  #pragma unroll 1
  for (int kt = 0; kt < NKT; ++kt) {
    const int cur = kt & 1;
    if (kt + 1 < NKT) fetch(kt + 1);
    wmma_step(As[cur], Bs[cur], am, bn, l16, ak0, bk0, acc);
    if (kt + 1 < NKT) stage(cur ^ 1);
    __syncthreads();
  }

  epilogue(Y, bias, scale, m0, n0, am, bn, lhalf, l16, acc);
}

extern "C" void kernel_launch(void* const* d_in, const int* in_sizes, int n_in,
                              void* d_out, int out_size, void* d_ws, size_t ws_size,
                              hipStream_t stream) {
  (void)in_sizes; (void)n_in; (void)out_size;
  const float* X    = (const float*)d_in[0];
  const float* W    = (const float*)d_in[1];
  const float* bias = (const float*)d_in[2];
  float* Y   = (float*)d_out;
  float* wsf = (float*)d_ws;
  float* scaleSlot = wsf;        // ws[0]
  float* partials  = wsf + 64;   // +256 B

  const size_t XB_OFF = 16384;
  const size_t XB_BYTES = (size_t)MDIM * KDIM * 2;   // 128 MB
  const size_t WQ_OFF = XB_OFF + XB_BYTES;
  const size_t WQ_BYTES = (size_t)NDIM * KDIM * 2;   // 32 MB
  const size_t NEED = WQ_OFF + WQ_BYTES;

  absum_partial<<<RB, 256, 0, stream>>>(W, partials);
  absum_final<<<1, 256, 0, stream>>>(partials, scaleSlot);

  dim3 grid(NDIM / BN, MDIM / BM);
  if (ws_size >= NEED) {
    unsigned short* Xb = (unsigned short*)((char*)d_ws + XB_OFF);
    unsigned short* Wq = (unsigned short*)((char*)d_ws + WQ_OFF);
    convert_x<<<(MDIM * (size_t)KDIM) / (8 * 256), 256, 0, stream>>>(X, Xb);
    quant_w<<<(NDIM * (size_t)KDIM) / (8 * 256), 256, 0, stream>>>(W, scaleSlot, Wq);
    bitlinear_gemm_pre<<<grid, 256, 0, stream>>>(Xb, Wq, bias, scaleSlot, Y);
  } else {
    bitlinear_gemm_otf<<<grid, 256, 0, stream>>>(X, W, bias, scaleSlot, Y);
  }
}
